// GlimpseModel_33414845563393
// MI455X (gfx1250) — matrix operand order repaired and checked by reading, not compiled, not run on hardware
//
#include <hip/hip_runtime.h>
#include <hip/hip_bf16.h>

typedef __attribute__((ext_vector_type(16))) _Float16 v16h;
typedef __attribute__((ext_vector_type(8)))  float    v8f;

#define Hh 100
#define Ww 100
#define Kk 144
#define Bb 32
#define NT 9            // 144 / 16 n-tiles
#define NORM_F (2.0f / 100.0f)
#define EPS_F 1e-7f

// LDS row strides in dwords (odd-ish to spread banks)
#define KXP_STRIDE 17   // kxP  [64][17]   packed f16 pairs, B-matrix layout
#define KYT_STRIDE 17   // kyT  [112][17]  f32, row = h, col = local k
#define IMG_STRIDE 65   // imgP [112][65]  packed f16 pairs of the image

__device__ inline unsigned pack2h(float a, float b) {
    union { _Float16 h[2]; unsigned u; } x;
    x.h[0] = (_Float16)a;   // even element -> [15:0]
    x.h[1] = (_Float16)b;   // odd  element -> [31:16]
    return x.u;
}

union FragU { unsigned u[8]; v16h h; };

__global__ __launch_bounds__(128)
void glimpse_wmma_kernel(const float* __restrict__ imgs,
                         const float* __restrict__ s_c,
                         const float* __restrict__ s_z,
                         const float* __restrict__ mu,
                         const float* __restrict__ sigma,
                         float* __restrict__ out)
{
    __shared__ float muxL[16], muyL[16], inv2L[16];
    __shared__ float sumxL[16], sumyL[16], numerL[16];
    __shared__ unsigned kxP [64  * KXP_STRIDE];
    __shared__ float    kyT [112 * KYT_STRIDE];
    __shared__ unsigned imgP[112 * IMG_STRIDE];

    const int tid = threadIdx.x;
    const int bid = blockIdx.x;
    const int b   = bid / NT;        // batch
    const int nt  = bid % NT;        // n-tile: local kernels k0..k0+15
    const int k0  = nt * 16;

    const float* img = imgs + b * (Hh * Ww);

    // Early HBM->L2 pull of this batch's image (gfx1250 global_prefetch_b8);
    // 128 threads x 312B stride covers the whole 40 KB image.
    __builtin_prefetch(img + tid * 78, 0, 0);

    // ---- per-kernel parameters (f32) ----
    if (tid < 16) {
        int k = k0 + tid;
        float sz = s_z[b];
        muxL[tid] = (s_c[2 * b + 0] + mu[2 * k + 0]) * sz;
        muyL[tid] = (s_c[2 * b + 1] + mu[2 * k + 1]) * sz;
        float sg  = sigma[k] * NORM_F * sz;
        inv2L[tid] = -0.5f / (sg * sg);
        sumxL[tid] = 0.f; sumyL[tid] = 0.f; numerL[tid] = 0.f;
    }
    __syncthreads();

    const float lstep = 2.0f / (float)(Hh - 1);   // lin[i] = -1 + i*lstep

    // ---- kx: f16 pairs in B-matrix friendly layout + f32 row sums ----
    for (int idx = tid; idx < 16 * 64; idx += 128) {
        int k = idx & 15, p = idx >> 4;           // p = w-pair index (w = 2p, 2p+1)
        int w0 = 2 * p, w1 = w0 + 1;
        float iv = inv2L[k], mx = muxL[k];
        float d0 = (-1.0f + lstep * (float)w0) - mx;
        float d1 = (-1.0f + lstep * (float)w1) - mx;
        float v0 = (w0 < Ww) ? __expf(iv * d0 * d0) : 0.f;
        float v1 = (w1 < Ww) ? __expf(iv * d1 * d1) : 0.f;
        atomicAdd(&sumxL[k], v0 + v1);            // zero padding contributes 0
        kxP[p * KXP_STRIDE + k] = pack2h(v0, v1);
    }

    // ---- ky: f32 (used in the post-GEMM contraction) + row sums ----
    for (int idx = tid; idx < 16 * 112; idx += 128) {
        int k = idx & 15, h = idx >> 4;
        float val = 0.f;
        if (h < Hh) {
            float d = (-1.0f + lstep * (float)h) - muyL[k];
            val = __expf(inv2L[k] * d * d);
            atomicAdd(&sumyL[k], val);
        }
        kyT[h * KYT_STRIDE + k] = val;            // rows 100..111 zeroed
    }

    // ---- image: f32 -> packed f16 pairs, zero padded to 112 x 128 ----
    for (int idx = tid; idx < 112 * 64; idx += 128) {
        int p = idx & 63, h = idx >> 6;
        float f0 = 0.f, f1 = 0.f;
        if (h < Hh && p < 50) {                   // 100 floats = exactly 50 float2
            float2 v = ((const float2*)(img + h * Ww))[p];
            f0 = v.x; f1 = v.y;
        }
        imgP[h * IMG_STRIDE + p] = pack2h(f0, f1);
    }
    __syncthreads();

    // ---- WMMA: P = imgs(112x128) @ kx^T(128x16), fused ky contraction ----
    const int lane = tid & 31;
    const int half = lane >> 4;
    const int ln16 = lane & 15;
    // wave id as an SGPR -> scalar branches, EXEC stays all-1s around WMMA
    const int swid = __builtin_amdgcn_readfirstlane(tid >> 5);

    // Preload all 4 B fragments (K = 0..127 over W), reused across m-tiles.
    // B 32x16 f16 as loaded here (contiguous k, same order as A): VGPR v -> K pair (v + 8*half), N = lane%16.
    FragU bf[4];
#pragma unroll
    for (int s = 0; s < 4; ++s)
#pragma unroll
        for (int v = 0; v < 8; ++v)
            bf[s].u[v] = kxP[(s * 16 + v + half * 8) * KXP_STRIDE + ln16];

    float partial = 0.f;
    auto do_mtile = [&](int mt) {
        v8f acc = {};
#pragma unroll
        for (int s = 0; s < 4; ++s) {
            // A 16x32 f16 as loaded here: both lane halves hold rows M=0..15;
            // VGPR v -> K pair s*16 + v + 8*half (contiguous k, same order as B).
            FragU af;
            int h = mt * 16 + ln16;
#pragma unroll
            for (int v = 0; v < 8; ++v) {
                int p = s * 16 + v + half * 8;
                af.u[v] = imgP[h * IMG_STRIDE + p];
            }
            acc = __builtin_amdgcn_wmma_f32_16x16x32_f16(
                      false, af.h, false, bf[s].h,
                      (short)0, acc, false, false);
        }
        // C layout: VGPR r -> row M = r + 8*half, col N = lane%16.
        // numer(k) += sum_h ky(k,h) * P(h,k)   (f32)
#pragma unroll
        for (int r = 0; r < 8; ++r) {
            int h = mt * 16 + r + half * 8;
            partial += acc[r] * kyT[h * KYT_STRIDE + ln16];
        }
    };

    // 7 m-tiles over 4 waves: wave w handles {w} and, for w<3, {w+4}.
    do_mtile(swid);
    if (swid < 3)            // scalar (SGPR) condition: s_cbranch, EXEC untouched
        do_mtile(swid + 4);

    // fold the two lane halves (same k, disjoint rows), then across waves
    partial += __shfl_xor(partial, 16, 32);
    if (half == 0) atomicAdd(&numerL[ln16], partial);
    __syncthreads();

    if (tid < 16) {
        float denom = sumyL[tid] * sumxL[tid] + EPS_F;
        out[b * Kk + k0 + tid] = numerL[tid] / denom;
    }
}

extern "C" void kernel_launch(void* const* d_in, const int* in_sizes, int n_in,
                              void* d_out, int out_size, void* d_ws, size_t ws_size,
                              hipStream_t stream) {
    (void)in_sizes; (void)n_in; (void)out_size; (void)d_ws; (void)ws_size;
    const float* imgs  = (const float*)d_in[0];   // (32,100,100) f32
    const float* s_c   = (const float*)d_in[1];   // (32,2)       f32
    const float* s_z   = (const float*)d_in[2];   // (32,1)       f32
    const float* mu    = (const float*)d_in[3];   // (144,2)      f32
    const float* sigma = (const float*)d_in[4];   // (144,)       f32
    float* out = (float*)d_out;                   // (32,144)     f32

    dim3 grid(Bb * NT);   // 288 independent workgroups
    dim3 block(128);      // 4 wave32s
    glimpse_wmma_kernel<<<grid, block, 0, stream>>>(imgs, s_c, s_z, mu, sigma, out);
}
